// RNN_23940147708270
// MI455X (gfx1250) — compile-verified, matrix operand order checked by reading, not run on hardware
//
#include <hip/hip_runtime.h>

#define B_ 64
#define T_ 512
#define D_ 512
#define H_ 512
#define G_ 2048  // 4*H

typedef __bf16 bf16;
typedef bf16 v16bf __attribute__((ext_vector_type(16)));
typedef bf16 v8bf  __attribute__((ext_vector_type(8)));
typedef float v8f  __attribute__((ext_vector_type(8)));

// ---------------------------------------------------------------------------
// Fragment loader. With A stored row-major [M,K] and B stored transposed
// [N,K] (both bf16, K-stride 512), A and B fragments have identical shape:
// lane (r<16): elements 0..7 = K 0..7, 8..15 = K 16..23 (row r)
// lane (r+16): elements 0..7 = K 8..15, 8..15 = K 24..31 (row r)
// p must already include +r*ld + ((lane>>4)<<3); two b128 loads per lane.
// ---------------------------------------------------------------------------
__device__ __forceinline__ v16bf frag16(const bf16* p) {
  const v8bf lo = *(const v8bf*)p;
  const v8bf hi = *(const v8bf*)(p + 16);
  v16bf f;
#pragma unroll
  for (int i = 0; i < 8; ++i) { f[i] = lo[i]; f[i + 8] = hi[i]; }
  return f;
}

// ---------------------------------------------------------------------------
// Elementwise fp32 -> bf16
// ---------------------------------------------------------------------------
__global__ void __launch_bounds__(256) f32_to_bf16_k(const float* __restrict__ in,
                                                     bf16* __restrict__ out, int n) {
  int i = blockIdx.x * blockDim.x + threadIdx.x;
  if (i < n) out[i] = (bf16)in[i];
}

// ---------------------------------------------------------------------------
// Weight transpose+convert: in [D_,G_] fp32 -> out [G_,D_] bf16 (coalesced out)
// ---------------------------------------------------------------------------
__global__ void __launch_bounds__(256) transpose_bf16_k(const float* __restrict__ in,
                                                        bf16* __restrict__ out) {
  int i = blockIdx.x * blockDim.x + threadIdx.x;  // i = g*D_ + d
  int g = i / D_;
  int d = i - g * D_;
  out[i] = (bf16)in[(size_t)d * G_ + g];
}

// ---------------------------------------------------------------------------
// Input projection GEMM: XZ[t*B+b, g] = sum_d A[b,t,d] * W[d,g] + bias[g]
// A is bf16 [B_,T_,*]; W pre-transposed bf16 [G_,D_]. Block tile 128x128,
// 8 waves, each wave 32x64 = 2x4 WMMA accumulators, K-loop of 16 x 32.
// ---------------------------------------------------------------------------
__global__ void __launch_bounds__(256) gemm_xz_k(const bf16* __restrict__ Abf,
                                                 const bf16* __restrict__ WT,
                                                 const float* __restrict__ bias,
                                                 float* __restrict__ XZ) {
  const int lane = threadIdx.x & 31;
  const int wave = threadIdx.x >> 5;
  const int wm = wave & 3;        // 32-row strip
  const int wn = wave >> 2;       // 64-col strip
  const int blockM = blockIdx.y * 128;
  const int blockN = blockIdx.x * 128;
  const int r15 = lane & 15;
  const int khalf = (lane >> 4) << 3;

  v8f acc[2][4];
#pragma unroll
  for (int mi = 0; mi < 2; ++mi)
#pragma unroll
    for (int ni = 0; ni < 4; ++ni)
#pragma unroll
      for (int e = 0; e < 8; ++e) acc[mi][ni][e] = 0.f;

  // Logical GEMM row r = t*B_ + b lives at memory row (b*T_ + t). 16-row
  // tiles never straddle a batch block (16 | 64), so t is tile-uniform.
  const bf16* aRow[2];
#pragma unroll
  for (int mi = 0; mi < 2; ++mi) {
    int row = blockM + wm * 32 + mi * 16 + r15;
    int t = row >> 6;
    int b = row & 63;
    aRow[mi] = Abf + ((size_t)b * T_ + t) * D_ + khalf;
  }
  const bf16* bRow[4];
#pragma unroll
  for (int ni = 0; ni < 4; ++ni) {
    int col = blockN + wn * 64 + ni * 16 + r15;
    bRow[ni] = WT + (size_t)col * D_ + khalf;
  }

  for (int k = 0; k < D_; k += 32) {
    v16bf af[2];
#pragma unroll
    for (int mi = 0; mi < 2; ++mi) af[mi] = frag16(aRow[mi] + k);
#pragma unroll
    for (int ni = 0; ni < 4; ++ni) {
      v16bf bfr = frag16(bRow[ni] + k);
#pragma unroll
      for (int mi = 0; mi < 2; ++mi)
        acc[mi][ni] = __builtin_amdgcn_wmma_f32_16x16x32_bf16(
            false, af[mi], false, bfr, (short)0, acc[mi][ni], false, false);
    }
  }

  // Epilogue: C layout — element e: row = tileM + (lane>>4)*8 + e, col = tileN + (lane&15)
#pragma unroll
  for (int ni = 0; ni < 4; ++ni) {
    int col = blockN + wn * 64 + ni * 16 + r15;
    float bv = bias[col];
#pragma unroll
    for (int mi = 0; mi < 2; ++mi) {
      int row0 = blockM + wm * 32 + mi * 16 + khalf;
#pragma unroll
      for (int e = 0; e < 8; ++e)
        XZ[(size_t)(row0 + e) * G_ + col] = acc[mi][ni][e] + bv;
    }
  }
}

// ---------------------------------------------------------------------------
// Persistent LSTM scan. Grid = 16 WGs; WG wg owns hidden block j0 = wg*32
// (all 4 gates for those units). Per step: z[64x128] = h[64x512] @ RKT-cols
// via WMMA, gate fusion through LDS, cell state in registers, h double-
// buffered in bf16 global, one atomic grid barrier per step.
// ---------------------------------------------------------------------------
__global__ void __launch_bounds__(256) lstm_scan_k(const float* __restrict__ XZ,
                                                   const bf16* __restrict__ RKT,
                                                   bf16* __restrict__ hbuf,   // [2][B_*H_]
                                                   int* __restrict__ ctr,     // [T_]
                                                   bf16* __restrict__ out_bf,
                                                   float* __restrict__ out_f32,
                                                   int nwg) {
  __shared__ float zsh[64 * 128];
  const int lane = threadIdx.x & 31;
  const int wave = threadIdx.x >> 5;
  const int mt = wave & 3;    // 16-row batch tile
  const int ng = wave >> 2;   // n-tile group: tiles ng*4 .. ng*4+3
  const int j0 = blockIdx.x * 32;
  const int r15 = lane & 15;
  const int khalf = (lane >> 4) << 3;

  // B rows fixed for all steps: local col n -> gate q = n>>5, unit jj = n&31,
  // global row of RKT = q*H_ + j0 + jj (contiguous in lane within a tile).
  const bf16* bRow[4];
#pragma unroll
  for (int ni = 0; ni < 4; ++ni) {
    int n = (ng * 4 + ni) * 16 + r15;
    int g = (n >> 5) * H_ + j0 + (n & 31);
    bRow[ni] = RKT + (size_t)g * D_ + khalf;
  }

  float creg[8];
#pragma unroll
  for (int r = 0; r < 8; ++r) creg[r] = 0.f;

  for (int t = 0; t < T_; ++t) {
    const bf16* hcur = hbuf + (size_t)(t & 1) * (B_ * H_);
    bf16* hnext = hbuf + (size_t)((t + 1) & 1) * (B_ * H_);

    v8f acc[4];
#pragma unroll
    for (int ni = 0; ni < 4; ++ni)
#pragma unroll
      for (int e = 0; e < 8; ++e) acc[ni][e] = 0.f;

    const bf16* aRow = hcur + (size_t)(mt * 16 + r15) * H_ + khalf;
    for (int k = 0; k < H_; k += 32) {
      v16bf af = frag16(aRow + k);
#pragma unroll
      for (int ni = 0; ni < 4; ++ni) {
        v16bf bfr = frag16(bRow[ni] + k);
        acc[ni] = __builtin_amdgcn_wmma_f32_16x16x32_bf16(
            false, af, false, bfr, (short)0, acc[ni], false, false);
      }
    }

    // z -> LDS for gate-fusion re-layout
#pragma unroll
    for (int ni = 0; ni < 4; ++ni) {
      int col = (ng * 4 + ni) * 16 + r15;
      int row0 = mt * 16 + khalf;
#pragma unroll
      for (int e = 0; e < 8; ++e) zsh[(row0 + e) * 128 + col] = acc[ni][e];
    }
    __syncthreads();

    // Elementwise: 64*32 = 2048 (b, jj) pairs, 8 per thread; c in registers.
#pragma unroll
    for (int r = 0; r < 8; ++r) {
      int idx = r * 256 + threadIdx.x;
      int b = idx >> 5;
      int jj = idx & 31;
      const float* xzp = XZ + ((size_t)t * B_ + b) * G_ + j0 + jj;
      float zi = zsh[b * 128 + jj] + xzp[0];
      float zf = zsh[b * 128 + 32 + jj] + xzp[H_];
      float zg = zsh[b * 128 + 64 + jj] + xzp[2 * H_];
      float zo = zsh[b * 128 + 96 + jj] + xzp[3 * H_];
      float ig = 1.f / (1.f + __expf(-zi));
      float fg = 1.f / (1.f + __expf(-zf));
      float gg = tanhf(zg);
      float og = 1.f / (1.f + __expf(-zo));
      float c = fg * creg[r] + ig * gg;
      creg[r] = c;
      float h = og * tanhf(c);
      hnext[b * H_ + j0 + jj] = (bf16)h;
      size_t oidx = ((size_t)b * T_ + t) * H_ + j0 + jj;
      if (out_bf) out_bf[oidx] = (bf16)h;
      if (out_f32) out_f32[oidx] = h;
    }

    // Grid-wide barrier: all WGs must publish h(t) before step t+1 reads it.
    __threadfence();
    __syncthreads();
    if (threadIdx.x == 0) {
      __hip_atomic_fetch_add(&ctr[t], 1, __ATOMIC_ACQ_REL, __HIP_MEMORY_SCOPE_AGENT);
      while (__hip_atomic_load(&ctr[t], __ATOMIC_ACQUIRE, __HIP_MEMORY_SCOPE_AGENT) < nwg) {
        __builtin_amdgcn_s_sleep(1);
      }
    }
    __syncthreads();
  }
}

// ---------------------------------------------------------------------------
// Workspace layout (bytes, all offsets 256-aligned)
// ---------------------------------------------------------------------------
static constexpr size_t XZ_OFF = 0;
static constexpr size_t XZ_BYTES = (size_t)T_ * B_ * G_ * 4;        // 256 MB
static constexpr size_t XBF_OFF = XZ_OFF + XZ_BYTES;
static constexpr size_t XBF_BYTES = (size_t)B_ * T_ * D_ * 2;       // 32 MB
static constexpr size_t YBF_OFF = XBF_OFF + XBF_BYTES;
static constexpr size_t YBF_BYTES = XBF_BYTES;                      // 32 MB
static constexpr size_t WT_OFF = YBF_OFF + YBF_BYTES;
static constexpr size_t WT_MAT = (size_t)G_ * D_ * 2;               // 2 MB each
static constexpr size_t HB_OFF = WT_OFF + 4 * WT_MAT;
static constexpr size_t HB_BYTES = 2 * (size_t)B_ * H_ * 2;         // 128 KB
static constexpr size_t CTR_OFF = HB_OFF + HB_BYTES;
static constexpr size_t CTR_BYTES = 2048 * sizeof(int);

extern "C" void kernel_launch(void* const* d_in, const int* in_sizes, int n_in,
                              void* d_out, int out_size, void* d_ws, size_t ws_size,
                              hipStream_t stream) {
  (void)in_sizes; (void)n_in; (void)out_size; (void)ws_size;

  const float* x = (const float*)d_in[0];     // [B,T,D]
  const float* kern = (const float*)d_in[1];  // [2,H,4H]
  const float* rk = (const float*)d_in[2];    // [2,H,4H]
  const float* bias = (const float*)d_in[3];  // [2,4H]

  char* ws = (char*)d_ws;
  float* XZ = (float*)(ws + XZ_OFF);
  bf16* XBF = (bf16*)(ws + XBF_OFF);
  bf16* YBF = (bf16*)(ws + YBF_OFF);
  bf16* K0T = (bf16*)(ws + WT_OFF + 0 * WT_MAT);
  bf16* R0T = (bf16*)(ws + WT_OFF + 1 * WT_MAT);
  bf16* K1T = (bf16*)(ws + WT_OFF + 2 * WT_MAT);
  bf16* R1T = (bf16*)(ws + WT_OFF + 3 * WT_MAT);
  bf16* HBUF = (bf16*)(ws + HB_OFF);
  int* CTR = (int*)(ws + CTR_OFF);

  dim3 tb(256);

  // Reset per-call state (graph-capture safe).
  hipMemsetAsync(HBUF, 0, HB_BYTES, stream);
  hipMemsetAsync(CTR, 0, CTR_BYTES, stream);

  // Weight convert + transpose to [4H, H] bf16 (one-time 8 MB; L2-resident after).
  int wgrid = (G_ * D_) / 256;
  transpose_bf16_k<<<wgrid, tb, 0, stream>>>(kern, K0T);
  transpose_bf16_k<<<wgrid, tb, 0, stream>>>(rk, R0T);
  transpose_bf16_k<<<wgrid, tb, 0, stream>>>(kern + (size_t)D_ * G_, K1T);
  transpose_bf16_k<<<wgrid, tb, 0, stream>>>(rk + (size_t)D_ * G_, R1T);

  // Activations to bf16.
  f32_to_bf16_k<<<(B_ * T_ * D_) / 256, tb, 0, stream>>>(x, XBF, B_ * T_ * D_);

  dim3 ggrid(G_ / 128, (T_ * B_) / 128);  // (16, 256)

  // Layer 0
  gemm_xz_k<<<ggrid, tb, 0, stream>>>(XBF, K0T, bias, XZ);
  lstm_scan_k<<<16, tb, 0, stream>>>(XZ, R0T, HBUF, CTR, YBF, nullptr, 16);

  // Layer 1
  hipMemsetAsync(HBUF, 0, HB_BYTES, stream);
  gemm_xz_k<<<ggrid, tb, 0, stream>>>(YBF, K1T, bias + G_, XZ);
  lstm_scan_k<<<16, tb, 0, stream>>>(XZ, R1T, HBUF, CTR + 1024, nullptr, (float*)d_out, 16);
}
